// SSMPhonotacticsModel_24773371364111
// MI455X (gfx1250) — matrix operand order repaired and checked by reading, not compiled, not run on hardware
//
#include <hip/hip_runtime.h>
#include <hip/hip_bf16.h>

#define XD   512
#define SD   128
#define BSZ  256
#define TLEN 256
#define MT   16          // batch rows per workgroup
#define KB   (XD / 32)   // 16 k-blocks of 32

#define AS1 __attribute__((address_space(1)))

typedef __attribute__((ext_vector_type(16))) __bf16       v16bf;
typedef __attribute__((ext_vector_type(8)))  float        v8f;
typedef __attribute__((ext_vector_type(4)))  unsigned int v4u;

union Frag {
    v16bf v;
    v4u   q[2];
};

__device__ __forceinline__ unsigned short f32_to_bf16_rne(float f) {
    unsigned int u = __builtin_bit_cast(unsigned int, f);
    u += 0x7fffu + ((u >> 16) & 1u);
    return (unsigned short)(u >> 16);
}

// ---------------------------------------------------------------------------
// Prep: swizzle A and Cpi=C*pi into WMMA B-operand fragment tiles (bf16),
// and transpose B -> Bt[s][n] for coalesced epilogue reads.
//
// Tile layout: block (nt, kt) is 1024 contiguous bf16 (2 KB); within a
// block, lane l element i holds Src[nt*16 + (l&15)][kt*32 + 16*(l>>4) + i]
// = a 32x16 (KxN) B fragment in contiguous k order (same as loadA), so the main
// loop issues two fully coalesced global_load_b128 per fragment.
// ---------------------------------------------------------------------------
__global__ void ssm_prep(const float* __restrict__ A,
                         const float* __restrict__ B,
                         const float* __restrict__ C,
                         const float* __restrict__ pi,
                         unsigned short* __restrict__ Atile,
                         unsigned short* __restrict__ Cpitile,
                         float* __restrict__ Bt) {
    const int e = blockIdx.x * blockDim.x + threadIdx.x;   // 0 .. 262143
    {
        const int blk = e >> 10, w = e & 1023;
        const int l = w >> 4, i = w & 15;
        const int nt = blk >> 4, kt = blk & 15;
        const int n = nt * 16 + (l & 15);
        const int k = kt * 32 + ((l >> 4) << 4) + i;
        if (e < XD * XD) Atile[e] = f32_to_bf16_rne(A[n * XD + k]);
    }
    if (e < SD * XD) {
        const int blk = e >> 10, w = e & 1023;
        const int l = w >> 4, i = w & 15;
        const int st = blk >> 4, kt = blk & 15;
        const int s = st * 16 + (l & 15);
        const int k = kt * 32 + ((l >> 4) << 4) + i;
        Cpitile[e] = f32_to_bf16_rne(C[s * XD + k] * pi[s * XD + k]);
        // B transpose: e = s*512 + n
        const int s2 = e >> 9, n2 = e & (XD - 1);
        Bt[e] = B[n2 * SD + s2];
    }
}

// ---------------------------------------------------------------------------
// Fused recurrence + emission + log-softmax + NLL accumulation.
// One workgroup owns 16 batch rows for all 256 timesteps.
// ---------------------------------------------------------------------------
__global__ __launch_bounds__(256)
void ssm_fused(const unsigned short* __restrict__ Atile,   // tiled bf16
               const unsigned short* __restrict__ Cpitile, // tiled bf16
               const float* __restrict__ Bt,               // [128][512] f32
               const float* __restrict__ pig,              // [128][512] f32
               const float* __restrict__ initg,            // [512]
               const int*   __restrict__ tokens,           // [256][256]
               float* __restrict__ out)                    // [256]
{
    __shared__ __align__(16) unsigned short xbf[MT][XD + 8];  // bf16 state
    __shared__ __align__(16) float          xf  [MT][XD + 4]; // f32 state
    __shared__ __align__(16) float          ybuf[MT][SD + 4]; // logits per step
    __shared__ int   tok[MT];
    __shared__ float llbuf[MT];

    const int tid  = threadIdx.x;
    const int wave = tid >> 5;         // 0..7
    const int lane = tid & 31;
    const int hl   = lane >> 4;
    const int l16  = lane & 15;
    const int b0   = blockIdx.x * MT;

    // ---- init state x = broadcast(init), ll = 0 ----
    for (int idx = tid; idx < MT * XD; idx += 256) {
        int m = idx >> 9, n = idx & (XD - 1);
        float v = initg[n];
        xf[m][n]  = v;
        xbf[m][n] = f32_to_bf16_rne(v);
    }
    if (tid < MT) llbuf[tid] = 0.0f;
    __syncthreads();

    for (int t = 0; t < TLEN; ++t) {
        // Launder the weight pointers (so LICM cannot hoist the fragment
        // loads out of the 256-step loop -> no scratch spills), on explicit
        // address-space(1) pointers so codegen emits saddr-form
        // global_load_b128 (LOADcnt only), not flat_load.
        const AS1 unsigned short* At = (const AS1 unsigned short*)Atile;
        const AS1 unsigned short* Ct = (const AS1 unsigned short*)Cpitile;
        asm volatile("" : "+s"(At), "+s"(Ct));

        if (tid < MT) tok[tid] = tokens[(b0 + tid) * TLEN + t];

        auto loadA = [&](int kt, Frag& a) {
            const unsigned short* p = &xbf[l16][kt * 32 + 16 * hl];
            a.q[0] = *(const v4u*)(p);
            a.q[1] = *(const v4u*)(p + 8);
        };
        auto loadB = [&](int kt, int j, Frag& b) {
            const int blk = (4 * wave + j) * KB + kt;
            const AS1 unsigned short* p = At + (blk << 10) + lane * 16;
            b.q[0] = *(const AS1 v4u*)(p);
            b.q[1] = *(const AS1 v4u*)(p + 8);
        };
        auto loadC = [&](int kt, Frag& c) {
            const int blk = wave * KB + kt;
            const AS1 unsigned short* p = Ct + (blk << 10) + lane * 16;
            c.q[0] = *(const AS1 v4u*)(p);
            c.q[1] = *(const AS1 v4u*)(p + 8);
        };

        v8f acc0 = {}, acc1 = {}, acc2 = {}, acc3 = {}, accY = {};

        // Software-pipelined k-loop: fragments for kt+1 are requested before
        // the kt WMMAs consume theirs, so every v_wmma sees partial loadcnt
        // waits instead of full drains.  Fully unrolled -> rotation is free.
        Frag fa, fb0, fb1, fb2, fb3, fc;
        loadA(0, fa);
        loadB(0, 0, fb0); loadB(0, 1, fb1); loadB(0, 2, fb2); loadB(0, 3, fb3);
        loadC(0, fc);

#pragma unroll
        for (int kt = 0; kt < KB; ++kt) {
            Frag na, nb0, nb1, nb2, nb3, nc;
            if (kt + 1 < KB) {
                loadA(kt + 1, na);
                loadB(kt + 1, 0, nb0); loadB(kt + 1, 1, nb1);
                loadB(kt + 1, 2, nb2); loadB(kt + 1, 3, nb3);
                loadC(kt + 1, nc);
            }
            acc0 = __builtin_amdgcn_wmma_f32_16x16x32_bf16(
                       false, fa.v, false, fb0.v, (short)0, acc0, false, false);
            acc1 = __builtin_amdgcn_wmma_f32_16x16x32_bf16(
                       false, fa.v, false, fb1.v, (short)0, acc1, false, false);
            acc2 = __builtin_amdgcn_wmma_f32_16x16x32_bf16(
                       false, fa.v, false, fb2.v, (short)0, acc2, false, false);
            acc3 = __builtin_amdgcn_wmma_f32_16x16x32_bf16(
                       false, fa.v, false, fb3.v, (short)0, acc3, false, false);
            accY = __builtin_amdgcn_wmma_f32_16x16x32_bf16(
                       false, fa.v, false, fc.v, (short)0, accY, false, false);
            fa = na; fb0 = nb0; fb1 = nb1; fb2 = nb2; fb3 = nb3; fc = nc;
        }

        // emission tile -> LDS (C/D layout: lane<16: N=lane,M=r; else M=r+8)
#pragma unroll
        for (int r = 0; r < 8; ++r)
            ybuf[r + 8 * hl][16 * wave + l16] = accY[r];

        __syncthreads();   // xbf reads done; ybuf complete; tok valid

        // ---- log-softmax + NLL (wave w: rows 2w / 2w+1, half-wave xor) ----
        {
            const int m = 2 * wave + hl;
            float vals[8];
            float mx = -1e30f;
#pragma unroll
            for (int q = 0; q < 8; ++q) {
                vals[q] = ybuf[m][l16 + 16 * q];
                mx = fmaxf(mx, vals[q]);
            }
#pragma unroll
            for (int d = 1; d < 16; d <<= 1)
                mx = fmaxf(mx, __shfl_xor(mx, d, 32));
            float sm = 0.0f;
#pragma unroll
            for (int q = 0; q < 8; ++q)
                sm += __expf(vals[q] - mx);
#pragma unroll
            for (int d = 1; d < 16; d <<= 1)
                sm += __shfl_xor(sm, d, 32);
            if (l16 == 0) {
                const int st = tok[m];
                llbuf[m] += ybuf[m][st] - mx - __logf(sm);
            }
        }

        // ---- mixing epilogue: x' = (1-p)x + p(xA^T + B[:,tok]) ----
#pragma unroll
        for (int j = 0; j < 4; ++j) {
            const v8f acc = (j == 0) ? acc0 : (j == 1) ? acc1 : (j == 2) ? acc2 : acc3;
            const int n = 64 * wave + 16 * j + l16;
#pragma unroll
            for (int r = 0; r < 8; ++r) {
                const int m = r + 8 * hl;
                const int s = tok[m];
                const float p    = pig[s * XD + n];   // coalesced over lanes 0-15
                const float bias = Bt[s * XD + n];    // coalesced (pre-transposed)
                const float xo   = xf[m][n];
                const float xn   = (1.0f - p) * xo + p * (acc[r] + bias);
                xf[m][n]  = xn;
                xbf[m][n] = f32_to_bf16_rne(xn);
            }
        }

        __syncthreads();   // x_new visible before next step's fragment loads
    }

    if (tid < MT) out[b0 + tid] = llbuf[tid];
}

// ---------------------------------------------------------------------------
extern "C" void kernel_launch(void* const* d_in, const int* in_sizes, int n_in,
                              void* d_out, int out_size, void* d_ws, size_t ws_size,
                              hipStream_t stream) {
    const float* A      = (const float*)d_in[0];   // [512,512]
    const float* B      = (const float*)d_in[1];   // [512,128]
    const float* C      = (const float*)d_in[2];   // [128,512]
    const float* pi     = (const float*)d_in[3];   // [128,512]
    const float* initv  = (const float*)d_in[4];   // [512]
    const int*   tokens = (const int*)d_in[5];     // [256,256]
    float*       out    = (float*)d_out;           // [256]

    unsigned char* ws = (unsigned char*)d_ws;
    unsigned short* Atile   = (unsigned short*)(ws);                    // 512 KB
    unsigned short* Cpitile = (unsigned short*)(ws + 524288);           // 128 KB
    float*          Btr     = (float*)(ws + 524288 + 131072);           // 256 KB

    ssm_prep<<<(XD * XD + 255) / 256, 256, 0, stream>>>(A, B, C, pi, Atile, Cpitile, Btr);
    ssm_fused<<<BSZ / MT, 256, 0, stream>>>(Atile, Cpitile, Btr, pi, initv, tokens, out);
}
